// LocGlbInteNBlks_LCGC_l1d2_34634616275329
// MI455X (gfx1250) — matrix operand order repaired and checked by reading, not compiled, not run on hardware
//
#include <hip/hip_runtime.h>
#include <hip/hip_bf16.h>

typedef __attribute__((ext_vector_type(16))) __bf16 v16bf;
typedef __attribute__((ext_vector_type(8)))  __bf16 v8bf;
typedef __attribute__((ext_vector_type(4)))  __bf16 v4bf;
typedef __attribute__((ext_vector_type(8)))  float  v8f;
typedef __attribute__((ext_vector_type(4)))  float  f4;

#define NG      8      // groups
#define NB      64     // batch
#define NC      512    // channels
#define ND      1024   // positions
#define CGRP    64     // NC / NG
#define SCALE_F 0.125f
#define EPS_F   1e-5f
#define LSTR    40     // LDS padded k-stride (bf16 elems): 80B rows -> conflict-free b128 reads

// ---- helpers ---------------------------------------------------------------

static __device__ __forceinline__ v16bf lds_read_B16(const __bf16* p) {
    // 16 contiguous bf16 (two 16B chunks, 16B aligned: 80*col + 32*half)
    v8bf lo = *(const v8bf*)(p);
    v8bf hi = *(const v8bf*)(p + 8);
    return __builtin_shufflevector(lo, hi, 0,1,2,3,4,5,6,7,8,9,10,11,12,13,14,15);
}

// A operand (16x32 bf16) from a row-major fp32 matrix row, contiguous layout:
//   e in [0,16): K = 16*half + e
static __device__ __forceinline__ v16bf load_A_row(const float* rowp, int half) {
    const float* p0 = rowp + 16 * half;
    const float* p1 = rowp + 16 * half + 8;
    f4 x0 = *(const f4*)(p0);
    f4 x1 = *(const f4*)(p0 + 4);
    f4 x2 = *(const f4*)(p1);
    f4 x3 = *(const f4*)(p1 + 4);
    v16bf a;
    a[0]=(__bf16)x0[0]; a[1]=(__bf16)x0[1]; a[2]=(__bf16)x0[2]; a[3]=(__bf16)x0[3];
    a[4]=(__bf16)x1[0]; a[5]=(__bf16)x1[1]; a[6]=(__bf16)x1[2]; a[7]=(__bf16)x1[3];
    a[8]=(__bf16)x2[0]; a[9]=(__bf16)x2[1]; a[10]=(__bf16)x2[2]; a[11]=(__bf16)x2[3];
    a[12]=(__bf16)x3[0]; a[13]=(__bf16)x3[1]; a[14]=(__bf16)x3[2]; a[15]=(__bf16)x3[3];
    return a;
}

static __device__ __forceinline__ float wave_reduce_add(float v) {
    #pragma unroll
    for (int off = 16; off >= 1; off >>= 1)
        v += __shfl_xor(v, off, 32);
    return v;
}

// ---- K0: zero small accumulators ------------------------------------------

__global__ void k0_zero(float* __restrict__ p, int n) {
    int i = blockIdx.x * 256 + threadIdx.x;
    if (i < n) p[i] = 0.0f;
}

// ---- K1: fused t/p/g projections + att reduction ---------------------------
// grid (ND/64, NC/128, NB), 256 threads (8 waves).
// Wave w: rows [blockIdx.y*128 + w*16, +16), cols [blockIdx.x*64, +64).

__global__ __launch_bounds__(256) void k1_proj(
    const float* __restrict__ Qp, const float* __restrict__ Kp, const float* __restrict__ Vp,
    const float* __restrict__ Wt, const float* __restrict__ Wp, const float* __restrict__ Wg,
    __bf16* __restrict__ t_ws, float* __restrict__ att)
{
    __shared__ __align__(16) __bf16 lds[3][64 * LSTR];

    const int b    = blockIdx.z;
    const int n0   = blockIdx.x * 64;
    const int tid  = threadIdx.x;
    const int lane = tid & 31;
    const int wave = tid >> 5;
    const int m0   = blockIdx.y * 128 + wave * 16;
    const int half = lane >> 4;
    const int l16  = lane & 15;

    const float* Xin[3] = {Vp, Qp, Kp};   // t<-V, p<-Q, g<-K
    const float* Wm[3]  = {Wt, Wp, Wg};

    v8f acc[3][4];
    #pragma unroll
    for (int gm = 0; gm < 3; ++gm)
        #pragma unroll
        for (int nt = 0; nt < 4; ++nt)
            acc[gm][nt] = (v8f){0,0,0,0,0,0,0,0};

    for (int k0 = 0; k0 < NC; k0 += 32) {
        __syncthreads();
        // stage 32x64 fp32 tiles of each input into LDS, transposed [n][k] as bf16
        #pragma unroll
        for (int gm = 0; gm < 3; ++gm) {
            const float* src = Xin[gm] + ((size_t)b * NC + k0) * ND + n0;
            if (k0 + 32 < NC) __builtin_prefetch(src + (size_t)32 * ND, 0, 0);
            #pragma unroll
            for (int it = 0; it < 2; ++it) {
                int idx4 = tid + it * 256;          // 0..511 float4s
                int kk   = idx4 >> 4;               // 0..31
                int nn   = (idx4 & 15) << 2;        // 0..60
                f4 vv = *(const f4*)(src + (size_t)kk * ND + nn);
                lds[gm][(nn + 0) * LSTR + kk] = (__bf16)vv[0];
                lds[gm][(nn + 1) * LSTR + kk] = (__bf16)vv[1];
                lds[gm][(nn + 2) * LSTR + kk] = (__bf16)vv[2];
                lds[gm][(nn + 3) * LSTR + kk] = (__bf16)vv[3];
            }
        }
        __syncthreads();

        v16bf Ar[3];
        #pragma unroll
        for (int gm = 0; gm < 3; ++gm)
            Ar[gm] = load_A_row(Wm[gm] + (size_t)(m0 + l16) * NC + k0, half);

        #pragma unroll
        for (int nt = 0; nt < 4; ++nt) {
            const int col = nt * 16 + l16;
            #pragma unroll
            for (int gm = 0; gm < 3; ++gm) {
                v16bf Bv = lds_read_B16(&lds[gm][col * LSTR + half * 16]);
                acc[gm][nt] = __builtin_amdgcn_wmma_f32_16x16x32_bf16(
                    false, Ar[gm], false, Bv, (short)0, acc[gm][nt], false, false);
            }
        }
    }

    // write t (bf16); accumulate p.g dot into att
    float part = 0.0f;
    #pragma unroll
    for (int nt = 0; nt < 4; ++nt) {
        #pragma unroll
        for (int i = 0; i < 8; ++i) {
            int row = m0 + i + half * 8;
            int col = n0 + nt * 16 + l16;
            t_ws[((size_t)b * NC + row) * ND + col] = (__bf16)acc[0][nt][i];
            part += acc[1][nt][i] * acc[2][nt][i];
        }
    }
    part = wave_reduce_add(part);
    if (lane == 0)
        atomicAdd(&att[b * NG + (m0 >> 6)], part);
}

// ---- K2: z = att * (Wz[g] x t), write z to out, accumulate GN stats --------
// grid (ND/64, NG, NB), 256 threads (8 waves).
// Wave w: rows (w&3)*16 of group, cols (w>>2)*32 of the 64-col block.

__global__ __launch_bounds__(256) void k2_zgemm(
    const __bf16* __restrict__ t_ws, const float* __restrict__ Wz,
    const float* __restrict__ att, float* __restrict__ zout,
    float* __restrict__ stats)
{
    __shared__ __align__(16) __bf16 lds[64 * LSTR];

    const int b    = blockIdx.z;
    const int g    = blockIdx.y;
    const int n0   = blockIdx.x * 64;
    const int tid  = threadIdx.x;
    const int lane = tid & 31;
    const int wave = tid >> 5;
    const int mt   = wave & 3;
    const int nh   = wave >> 2;
    const int half = lane >> 4;
    const int l16  = lane & 15;
    const int m0   = mt * 16;

    v8f acc[2] = { (v8f){0,0,0,0,0,0,0,0}, (v8f){0,0,0,0,0,0,0,0} };

    #pragma unroll
    for (int k0 = 0; k0 < CGRP; k0 += 32) {
        __syncthreads();
        const __bf16* src = t_ws + ((size_t)b * NC + g * CGRP + k0) * ND + n0;
        #pragma unroll
        for (int it = 0; it < 2; ++it) {
            int idx4 = tid + it * 256;
            int kk   = idx4 >> 4;
            int nn   = (idx4 & 15) << 2;
            v4bf vv = *(const v4bf*)(src + (size_t)kk * ND + nn);
            lds[(nn + 0) * LSTR + kk] = vv[0];
            lds[(nn + 1) * LSTR + kk] = vv[1];
            lds[(nn + 2) * LSTR + kk] = vv[2];
            lds[(nn + 3) * LSTR + kk] = vv[3];
        }
        __syncthreads();

        v16bf A = load_A_row(Wz + ((size_t)g * CGRP + m0 + l16) * CGRP + k0, half);
        #pragma unroll
        for (int nt2 = 0; nt2 < 2; ++nt2) {
            int col = (nh * 2 + nt2) * 16 + l16;
            v16bf Bv = lds_read_B16(&lds[col * LSTR + half * 16]);
            acc[nt2] = __builtin_amdgcn_wmma_f32_16x16x32_bf16(
                false, A, false, Bv, (short)0, acc[nt2], false, false);
        }
    }

    const float attv = att[b * NG + g] * SCALE_F;
    float s1 = 0.0f, s2 = 0.0f;
    #pragma unroll
    for (int nt2 = 0; nt2 < 2; ++nt2) {
        #pragma unroll
        for (int i = 0; i < 8; ++i) {
            float z = attv * acc[nt2][i];
            int row = g * CGRP + m0 + i + half * 8;
            int col = n0 + (nh * 2 + nt2) * 16 + l16;
            zout[((size_t)b * NC + row) * ND + col] = z;
            s1 += z;
            s2 += z * z;
        }
    }
    s1 = wave_reduce_add(s1);
    s2 = wave_reduce_add(s2);
    if (lane == 0) {
        atomicAdd(&stats[(b * NG + g) * 2 + 0], s1);
        atomicAdd(&stats[(b * NG + g) * 2 + 1], s2);
    }
}

// ---- K3: in-place GroupNorm + affine + residual ----------------------------

__global__ __launch_bounds__(256) void k3_norm(
    float* __restrict__ out, const float* __restrict__ Vp,
    const float* __restrict__ stats,
    const float* __restrict__ gamma, const float* __restrict__ beta)
{
    size_t i4   = (size_t)blockIdx.x * 256 + threadIdx.x;   // float4 index
    size_t flat = i4 * 4;
    int c = (int)((flat / ND) % NC);
    int b = (int)(flat / ((size_t)ND * NC));
    int g = c >> 6;

    const float invN = 1.0f / ((float)CGRP * (float)ND);
    float s1   = stats[(b * NG + g) * 2 + 0];
    float s2   = stats[(b * NG + g) * 2 + 1];
    float mean = s1 * invN;
    float var  = s2 * invN - mean * mean;
    float rs   = rsqrtf(var + EPS_F);
    float sc   = gamma[c] * rs;
    float off  = beta[c] - mean * sc;

    f4 z = ((const f4*)out)[i4];
    f4 v = ((const f4*)Vp)[i4];
    f4 r;
    #pragma unroll
    for (int j = 0; j < 4; ++j)
        r[j] = z[j] * sc + off + v[j];   // RESID = 1.0
    ((f4*)out)[i4] = r;
}

// ---- launcher ---------------------------------------------------------------

extern "C" void kernel_launch(void* const* d_in, const int* in_sizes, int n_in,
                              void* d_out, int out_size, void* d_ws, size_t ws_size,
                              hipStream_t stream) {
    const float* Qp    = (const float*)d_in[0];
    const float* Kp    = (const float*)d_in[1];
    const float* Vp    = (const float*)d_in[2];
    const float* Wt    = (const float*)d_in[3];
    const float* Wp    = (const float*)d_in[4];
    const float* Wg    = (const float*)d_in[5];
    const float* Wz    = (const float*)d_in[6];
    const float* gamma = (const float*)d_in[7];
    const float* beta  = (const float*)d_in[8];
    float* out = (float*)d_out;

    float*  att   = (float*)d_ws;                       // NB*NG floats
    float*  stats = att + NB * NG;                      // NB*NG*2 floats
    __bf16* t_ws  = (__bf16*)((char*)d_ws + 8192);      // NB*NC*ND bf16 (64 MB)

    k0_zero<<<dim3(6), dim3(256), 0, stream>>>(att, NB * NG * 3);
    k1_proj<<<dim3(ND / 64, NC / 128, NB), dim3(256), 0, stream>>>(
        Qp, Kp, Vp, Wt, Wp, Wg, t_ws, att);
    k2_zgemm<<<dim3(ND / 64, NG, NB), dim3(256), 0, stream>>>(
        t_ws, Wz, att, out, stats);
    k3_norm<<<dim3((NB * NC * (ND / 4)) / 256), dim3(256), 0, stream>>>(
        out, Vp, stats, gamma, beta);
}
